// AttentionBlock_47854525612313
// MI455X (gfx1250) — compile-verified, matrix operand order checked by reading, not run on hardware
//
#include <hip/hip_runtime.h>
#include <hip/hip_bf16.h>

// ---------------------------------------------------------------------------
// AttentionBlock for MI455X (gfx1250): GroupNorm -> QKV GEMM -> flash
// attention -> output GEMM + residual.  All matmuls run on
// v_wmma_f32_16x16x32_bf16 (fp32 accum, bf16 operands).
//
// All WMMA operands are staged so each lane's 16 fragment elements are two
// contiguous, 16B-aligned chunks (A-layout K-runs): B-side tiles are stored
// TRANSPOSED in LDS, so every fragment load is 2x ds_load_b128.
// Next-tile global_prefetch_b8 overlaps HBM/L2 latency with the WMMA burst.
// ---------------------------------------------------------------------------

typedef __bf16 bf16_t;
typedef __attribute__((ext_vector_type(16))) __bf16 v16bf;
typedef __attribute__((ext_vector_type(8)))  __bf16 v8bf;
typedef __attribute__((ext_vector_type(8)))  float  v8f;

// Fragment load for 16-bit 16x32 A/B layout (ISA 7.12.2):
// lane half = lane/16; element e -> k = (e/8)*16 + half*8 + (e%8).
// p points at the first element of this lane's 32-element K-run.
__device__ __forceinline__ v16bf load_frag(const bf16_t* p, int half) {
    const v8bf lo = *(const v8bf*)(p + (half << 3));        // k = half*8 ..
    const v8bf hi = *(const v8bf*)(p + 16 + (half << 3));   // k = 16+half*8 ..
    return __builtin_shufflevector(lo, hi, 0, 1, 2, 3, 4, 5, 6, 7,
                                   8, 9, 10, 11, 12, 13, 14, 15);
}

__device__ __forceinline__ v8f wmma_bf16(v16bf a, v16bf b, v8f c) {
    // (neg_a, A, neg_b, B, c_mod, C, reuse_a, reuse_b)
    return __builtin_amdgcn_wmma_f32_16x16x32_bf16(false, a, false, b,
                                                   (short)0, c, false, false);
}

// ---------------------------------------------------------------------------
// Kernel 1: GroupNorm.  x:[B,C,H,W] fp32 -> h:[B*S, C] bf16 (GEMM layout).
// ---------------------------------------------------------------------------
__global__ __launch_bounds__(256) void gn_kernel(
    const float* __restrict__ x, const float* __restrict__ gamma,
    const float* __restrict__ beta, bf16_t* __restrict__ h) {
    __shared__ float red[256];
    __shared__ float red2[256];
    const int bg = blockIdx.x;
    const int b = bg >> 5, g = bg & 31;
    const int c0 = g * 8;
    const size_t base = ((size_t)b * 256 + c0) * 4096;

    float sum = 0.f, sq = 0.f;
    for (int i = threadIdx.x; i < 8 * 4096 / 4; i += 256) {
        const float4 v4 = *(const float4*)(x + base + i * 4);
        sum += v4.x + v4.y + v4.z + v4.w;
        sq += v4.x * v4.x + v4.y * v4.y + v4.z * v4.z + v4.w * v4.w;
    }
    red[threadIdx.x] = sum;
    red2[threadIdx.x] = sq;
    __syncthreads();
    for (int off = 128; off > 0; off >>= 1) {
        if (threadIdx.x < (unsigned)off) {
            red[threadIdx.x]  += red[threadIdx.x + off];
            red2[threadIdx.x] += red2[threadIdx.x + off];
        }
        __syncthreads();
    }
    const float inv_n = 1.0f / 32768.0f;
    const float mean = red[0] * inv_n;
    const float var  = red2[0] * inv_n - mean * mean;
    const float rsq  = rsqrtf(var + 1e-5f);

    for (int i = threadIdx.x; i < 8 * 4096; i += 256) {
        const int c = c0 + (i >> 12);
        const int s = i & 4095;
        const float v = (x[base + i] - mean) * rsq * gamma[c] + beta[c];
        h[((size_t)b * 4096 + s) * 256 + c] = (bf16_t)v;
    }
}

// ---------------------------------------------------------------------------
// Shared GEMM body: C[64x64] = A[64xK]bf16 * W[KxN]fp32 (+bias).
// sBT holds the weight tile TRANSPOSED: sBT[n][k].
// ---------------------------------------------------------------------------
template <int K, int N>
__device__ __forceinline__ void gemm_tile(
    const bf16_t* __restrict__ A, const float* __restrict__ W,
    int m0, int n0, int tid, v8f acc[2][2]) {
    __shared__ bf16_t sA[64][40];   // sA[m][k]   (pitch 40 -> rows 16B aligned)
    __shared__ bf16_t sBT[64][40];  // sBT[n][k]  (transposed weight tile)

    const int lane = tid & 31, warp = tid >> 5;
    const int half = lane >> 4, l16 = lane & 15;
    const int warpM = warp >> 1, warpN = warp & 1;

    for (int kk = 0; kk < K; kk += 32) {
        __syncthreads();
        // A: 64x32 bf16, vectorized 16B copies.
        for (int i = tid; i < 64 * 32 / 8; i += 128) {
            const int r = i >> 2, c8 = (i & 3) * 8;
            *(v8bf*)(&sA[r][c8]) =
                *(const v8bf*)(A + (size_t)(m0 + r) * K + kk + c8);
        }
        // W: 32x64 fp32, float4 loads, convert, transposed LDS stores.
        for (int i = tid; i < 32 * 64 / 4; i += 128) {
            const int r = i >> 4, c4 = (i & 15) * 4;
            const float4 w4 = *(const float4*)(W + (size_t)(kk + r) * N + n0 + c4);
            sBT[c4 + 0][r] = (bf16_t)w4.x;
            sBT[c4 + 1][r] = (bf16_t)w4.y;
            sBT[c4 + 2][r] = (bf16_t)w4.z;
            sBT[c4 + 3][r] = (bf16_t)w4.w;
        }
        // Prefetch next K-step tiles into cache while WMMAs run.
        if (kk + 32 < K) {
            const int r = tid >> 1, c = (tid & 1) * 16;
            __builtin_prefetch(A + (size_t)(m0 + r) * K + kk + 32 + c, 0, 0);
            if (tid < 32)
                __builtin_prefetch(W + (size_t)(kk + 32 + tid) * N + n0, 0, 0);
        }
        __syncthreads();

        v16bf a[2], bfrag[2];
        for (int mt = 0; mt < 2; ++mt)
            a[mt] = load_frag(&sA[warpM * 32 + mt * 16 + l16][0], half);
        for (int nt = 0; nt < 2; ++nt)
            bfrag[nt] = load_frag(&sBT[warpN * 32 + nt * 16 + l16][0], half);
        for (int mt = 0; mt < 2; ++mt)
            for (int nt = 0; nt < 2; ++nt)
                acc[mt][nt] = wmma_bf16(a[mt], bfrag[nt], acc[mt][nt]);
    }
}

// ---------------------------------------------------------------------------
// Kernel 2: QKV GEMM.  [16384,256]bf16 x [256,768]fp32 -> bf16 [16384,768].
// ---------------------------------------------------------------------------
__global__ __launch_bounds__(128) void qkv_gemm_kernel(
    const bf16_t* __restrict__ A, const float* __restrict__ W,
    const float* __restrict__ bias, bf16_t* __restrict__ out) {
    constexpr int K = 256, N = 768;
    const int tid = threadIdx.x;
    const int lane = tid & 31, warp = tid >> 5;
    const int half = lane >> 4, l16 = lane & 15;
    const int warpM = warp >> 1, warpN = warp & 1;
    const int m0 = blockIdx.x * 64, n0 = blockIdx.y * 64;

    v8f acc[2][2] = {};
    gemm_tile<K, N>(A, W, m0, n0, tid, acc);

    for (int mt = 0; mt < 2; ++mt)
        for (int nt = 0; nt < 2; ++nt)
            for (int e = 0; e < 8; ++e) {
                const int m = m0 + warpM * 32 + mt * 16 + e + 8 * half;
                const int n = n0 + warpN * 32 + nt * 16 + l16;
                out[(size_t)m * N + n] = (bf16_t)(acc[mt][nt][e] + bias[n]);
            }
}

// ---------------------------------------------------------------------------
// Kernel 3: flash attention.  One block per (b, head, 64-query tile).
// K tile row-major (QK^T contraction over d -> contiguous); V tile stored
// TRANSPOSED (PV contraction over key -> contiguous).  P round-trips through
// a per-wave LDS strip to convert C-layout -> A-layout.
// ---------------------------------------------------------------------------
__global__ __launch_bounds__(128) void attn_kernel(
    const bf16_t* __restrict__ qkv, bf16_t* __restrict__ res) {
    constexpr int S = 4096;
    const float scale = 0.125f;  // D_K^-0.5

    __shared__ bf16_t sK[64][72];       // sK[key][d]
    __shared__ bf16_t sVT[64][72];      // sVT[d][key]  (transposed)
    __shared__ bf16_t sP[4][16][72];    // per-wave P scratch, sP[w][m][key]

    const int tid = threadIdx.x;
    const int lane = tid & 31, warp = tid >> 5;
    const int half = lane >> 4, l16 = lane & 15;
    const int bh = blockIdx.y;
    const int b = bh >> 2, head = bh & 3;
    const size_t rowbase = (size_t)b * S;
    const int q0 = blockIdx.x * 64 + warp * 16;
    const int hcol = head * 192;  // q:+0  k:+64  v:+128

    // Q fragments straight from global (16B-aligned chunks).
    v16bf qf[2];
    {
        const bf16_t* qrow = qkv + (rowbase + q0 + l16) * 768 + hcol;
        qf[0] = load_frag(qrow, half);
        qf[1] = load_frag(qrow + 32, half);
    }

    v8f acc[4] = {};
    float mrow[8], lrow[8];
    for (int e = 0; e < 8; ++e) { mrow[e] = -1e30f; lrow[e] = 0.f; }

    for (int kt = 0; kt < S / 64; ++kt) {
        __syncthreads();
        // Stage K (row-major) and V (transposed): 16B global loads.
        for (int i = tid; i < 64 * 64 / 8; i += 128) {
            const int r = i >> 3, c8 = (i & 7) * 8;
            const size_t row = (rowbase + kt * 64 + r) * 768;
            *(v8bf*)(&sK[r][c8]) = *(const v8bf*)(qkv + row + hcol + 64 + c8);
            const v8bf vv = *(const v8bf*)(qkv + row + hcol + 128 + c8);
            for (int j = 0; j < 8; ++j) sVT[c8 + j][r] = vv[j];
        }
        // Prefetch next tile's K/V rows (global_prefetch_b8) to hide latency
        // behind this tile's softmax + WMMA work.
        if (kt + 1 < S / 64) {
            const int r = tid >> 1;
            const size_t nrow = (rowbase + (kt + 1) * 64 + r) * 768;
            __builtin_prefetch(qkv + nrow + hcol + 64 + (tid & 1) * 64, 0, 0);
        }
        __syncthreads();

        // S = Q K^T : 16 queries x 64 keys (contraction over d).
        v8f sfr[4] = {};
        for (int nt = 0; nt < 4; ++nt)
            for (int kc = 0; kc < 2; ++kc) {
                const v16bf bk = load_frag(&sK[nt * 16 + l16][kc * 32], half);
                sfr[nt] = wmma_bf16(qf[kc], bk, sfr[nt]);
            }

        // Online softmax: row m = e + 8*half, columns striped across the
        // 16-lane half -> shfl_xor reductions (masks 1..8 stay in-half).
        for (int e = 0; e < 8; ++e) {
            float mx = -1e30f;
            for (int nt = 0; nt < 4; ++nt) {
                sfr[nt][e] *= scale;
                mx = fmaxf(mx, sfr[nt][e]);
            }
            for (int off = 1; off < 16; off <<= 1)
                mx = fmaxf(mx, __shfl_xor(mx, off, 32));
            const float newmax = fmaxf(mrow[e], mx);
            const float corr = __expf(mrow[e] - newmax);
            mrow[e] = newmax;
            float rs = 0.f;
            for (int nt = 0; nt < 4; ++nt) {
                const float p = __expf(sfr[nt][e] - newmax);
                sfr[nt][e] = p;
                rs += p;
            }
            for (int off = 1; off < 16; off <<= 1)
                rs += __shfl_xor(rs, off, 32);
            lrow[e] = lrow[e] * corr + rs;
            for (int nt = 0; nt < 4; ++nt) acc[nt][e] *= corr;
        }

        // C-layout -> A-layout for P via this wave's private LDS strip
        // (DS ops are in-order within a wave).
        for (int nt = 0; nt < 4; ++nt)
            for (int e = 0; e < 8; ++e)
                sP[warp][e + 8 * half][nt * 16 + l16] = (bf16_t)sfr[nt][e];
        v16bf pa[2];
        pa[0] = load_frag(&sP[warp][l16][0], half);
        pa[1] = load_frag(&sP[warp][l16][32], half);

        // O += P V : contraction over key (contiguous in sVT rows).
        for (int kc = 0; kc < 2; ++kc)
            for (int nt = 0; nt < 4; ++nt) {
                const v16bf bv = load_frag(&sVT[nt * 16 + l16][kc * 32], half);
                acc[nt] = wmma_bf16(pa[kc], bv, acc[nt]);
            }
    }

    for (int e = 0; e < 8; ++e) {
        const float inv = 1.0f / lrow[e];
        for (int nt = 0; nt < 4; ++nt) acc[nt][e] *= inv;
    }
    // res: [B*S, 256], column = head*64 + d.
    for (int nt = 0; nt < 4; ++nt)
        for (int e = 0; e < 8; ++e) {
            const int m = q0 + e + 8 * half;
            const int d = nt * 16 + l16;
            res[(rowbase + m) * 256 + head * 64 + d] = (bf16_t)acc[nt][e];
        }
}

// ---------------------------------------------------------------------------
// Kernel 4: output GEMM [16384,256] x [256,256] + bias + residual,
// stored fp32 transposed into [B,C,H,W].
// ---------------------------------------------------------------------------
__global__ __launch_bounds__(128) void out_gemm_kernel(
    const bf16_t* __restrict__ A, const float* __restrict__ W,
    const float* __restrict__ bias, const float* __restrict__ x,
    float* __restrict__ out) {
    constexpr int K = 256, N = 256;
    const int tid = threadIdx.x;
    const int lane = tid & 31, warp = tid >> 5;
    const int half = lane >> 4, l16 = lane & 15;
    const int warpM = warp >> 1, warpN = warp & 1;
    const int m0 = blockIdx.x * 64, n0 = blockIdx.y * 64;

    v8f acc[2][2] = {};
    gemm_tile<K, N>(A, W, m0, n0, tid, acc);

    // out[b][n][s] = acc + bias[n] + x[b][n][s]
    for (int mt = 0; mt < 2; ++mt)
        for (int nt = 0; nt < 2; ++nt)
            for (int e = 0; e < 8; ++e) {
                const int m = m0 + warpM * 32 + mt * 16 + e + 8 * half;
                const int n = n0 + warpN * 32 + nt * 16 + l16;
                const int b = m >> 12, s = m & 4095;
                const size_t idx = ((size_t)(b * 256 + n)) * 4096 + s;
                out[idx] = acc[mt][nt][e] + bias[n] + x[idx];
            }
}

// ---------------------------------------------------------------------------
extern "C" void kernel_launch(void* const* d_in, const int* in_sizes, int n_in,
                              void* d_out, int out_size, void* d_ws,
                              size_t ws_size, hipStream_t stream) {
    (void)in_sizes; (void)n_in; (void)out_size; (void)ws_size;
    const float* x     = (const float*)d_in[0];
    const float* gamma = (const float*)d_in[1];
    const float* beta  = (const float*)d_in[2];
    const float* w_qkv = (const float*)d_in[3];
    const float* b_qkv = (const float*)d_in[4];
    const float* w_out = (const float*)d_in[5];
    const float* b_out = (const float*)d_in[6];
    float* out = (float*)d_out;

    // workspace: h 8MB | qkv 24MB | attn-out 8MB  (40MB total)
    char* ws = (char*)d_ws;
    bf16_t* h   = (bf16_t*)(ws);
    bf16_t* qkv = (bf16_t*)(ws + 8388608);             // 16384*256*2
    bf16_t* res = (bf16_t*)(ws + 8388608 + 25165824);  // + 16384*768*2

    gn_kernel<<<dim3(128), 256, 0, stream>>>(x, gamma, beta, h);
    qkv_gemm_kernel<<<dim3(256, 12), 128, 0, stream>>>(h, w_qkv, b_qkv, qkv);
    attn_kernel<<<dim3(64, 16), 128, 0, stream>>>(qkv, res);
    out_gemm_kernel<<<dim3(256, 4), 128, 0, stream>>>(res, w_out, b_out, x, out);
}
